// Attention_2946347565533
// MI455X (gfx1250) — matrix operand order repaired and checked by reading, not compiled, not run on hardware
//
#include <hip/hip_runtime.h>

// ---------------------------------------------------------------------------
// Causal attention block for MI455X (gfx1250, wave32, WMMA).
// fp32 inputs -> bf16 compute with fp32 accumulation on v_wmma_f32_16x16x32_bf16.
// Round 4: fix async builtin pointer types — clang's diagnostic revealed the
// signature takes (v4i addrspace(1)*, v4i addrspace(3)*, imm, imm).
// ---------------------------------------------------------------------------

typedef unsigned short u16;
typedef unsigned int   u32;

constexpr int B_  = 4;
constexpr int S_  = 2048;
constexpr int D_  = 2048;
constexpr int H_  = 16;
constexpr int DH_ = 128;

typedef __attribute__((ext_vector_type(16))) __bf16 v16bf;
typedef __attribute__((ext_vector_type(8)))  float  v8f;
typedef int v4i __attribute__((vector_size(16)));

union Frag16 {            // 16 bf16 elements = one WMMA A/B operand per lane
    v16bf v;
    uint4 q[2];
};

__device__ __forceinline__ u16 f2bf(float f) {
    u32 u = __float_as_uint(f);
    u += 0x7FFFu + ((u >> 16) & 1u);       // round to nearest even
    return (u16)(u >> 16);
}
__device__ __forceinline__ float bf2f(u16 h) {
    return __uint_as_float((u32)h << 16);
}

// ---- gfx1250 async global->LDS copy (ASYNCcnt) with synchronous fallback ----
#if __has_builtin(__builtin_amdgcn_global_load_async_to_lds_b128) && \
    __has_builtin(__builtin_amdgcn_s_wait_asynccnt)
#define ASYNC_LDS 1
#endif

__device__ __forceinline__ void cp16_g2l(u16* lds_dst, const u16* gsrc) {
#ifdef ASYNC_LDS
    __builtin_amdgcn_global_load_async_to_lds_b128(
        (__attribute__((address_space(1))) v4i*)gsrc,
        (__attribute__((address_space(3))) v4i*)lds_dst,
        /*offset=*/0, /*cpol=*/0);
#else
    *(uint4*)lds_dst = *(const uint4*)gsrc;
#endif
}

template <int N>
__device__ __forceinline__ void wait_async() {
#ifdef ASYNC_LDS
    __builtin_amdgcn_s_wait_asynccnt(N);
#endif
}

// ---------------------------------------------------------------------------
// fp32 -> bf16 bulk convert
// ---------------------------------------------------------------------------
__global__ void f32_to_bf16_kernel(const float* __restrict__ in,
                                   u16* __restrict__ out, size_t n) {
    size_t i      = (size_t)blockIdx.x * blockDim.x + threadIdx.x;
    size_t stride = (size_t)gridDim.x * blockDim.x;
    for (; i < n; i += stride) out[i] = f2bf(in[i]);
}

// ---------------------------------------------------------------------------
// Generic bf16 GEMM:  C[M,N] = A[M,K] * Bw[N,K]^T  (Bw row-major = B columns)
// One wave computes a 16x64 tile (4 C fragments), K loop step 32.
// MODE 0: store fp32 C.   MODE 1: QKV epilogue scatter (q/k [B,H,S,DH] bf16,
//                                  v transposed [B,H,DH,S] bf16).
// ---------------------------------------------------------------------------
template <int MODE>
__global__ void __launch_bounds__(256)
gemm_bf16_kernel(const u16* __restrict__ A,
                 const u16* __restrict__ Bw,
                 int M, int N, int K,
                 float* __restrict__ C,
                 u16* __restrict__ qd,
                 u16* __restrict__ kd,
                 u16* __restrict__ vd) {
    const int lane = threadIdx.x & 31;
    const int wave = blockIdx.x * (blockDim.x >> 5) + (threadIdx.x >> 5);
    const int ntiles = N >> 6;
    const int mt = wave / ntiles;
    const int nt = wave - mt * ntiles;
    const int m0 = mt << 4, n0 = nt << 6;
    if (m0 >= M) return;

    const int lh = lane >> 4;   // lane half: 0/1
    const int ll = lane & 15;

    const u16* pa = A + (size_t)(m0 + ll) * K + lh * 8;

    v8f acc[4] = {};

    for (int k0 = 0; k0 < K; k0 += 32) {
        Frag16 a;
        a.q[0] = *(const uint4*)(pa + k0);        // K chunk {0..7}+8*lh
        a.q[1] = *(const uint4*)(pa + k0 + 16);   // K chunk {16..23}+8*lh
#pragma unroll
        for (int t = 0; t < 4; ++t) {
            const u16* pb = Bw + (size_t)(n0 + t * 16 + ll) * K + k0 + lh * 8;
            Frag16 b;
            b.q[0] = *(const uint4*)(pb);
            b.q[1] = *(const uint4*)(pb + 16);
            acc[t] = __builtin_amdgcn_wmma_f32_16x16x32_bf16(
                false, a.v, false, b.v, (short)0, acc[t], false, false);
        }
    }

#pragma unroll
    for (int t = 0; t < 4; ++t) {
        const int n = n0 + t * 16 + ll;
        if (MODE == 0) {
#pragma unroll
            for (int j = 0; j < 8; ++j) {
                const int m = m0 + j + lh * 8;
                C[(size_t)m * N + n] = acc[t][j];
            }
        } else {
            const int which = n / D_;            // 0=q 1=k 2=v (uniform per lane)
            const int r     = n - which * D_;
            const int h     = r / DH_;
            const int dh    = r - h * DH_;
#pragma unroll
            for (int j = 0; j < 8; ++j) {
                const int m = m0 + j + lh * 8;
                const int b = m / S_;
                const int s = m - b * S_;
                const size_t bh = (size_t)b * H_ + h;
                const u16 val = f2bf(acc[t][j]);
                if (which == 0)      qd[(bh * S_ + s) * DH_ + dh] = val;
                else if (which == 1) kd[(bh * S_ + s) * DH_ + dh] = val;
                else                 vd[(bh * DH_ + dh) * S_ + s] = val;   // V^T
            }
        }
    }
}

// ---------------------------------------------------------------------------
// RoPE on q and k (bf16 [B,H,S,DH]); folds softmax scale 1/sqrt(DH) into q.
// One thread per (row, freq-pair j<64): owns elements j and j+64 -> race-free.
// ---------------------------------------------------------------------------
__global__ void __launch_bounds__(256)
rope_kernel(u16* __restrict__ qd, u16* __restrict__ kd) {
    const size_t idx = (size_t)blockIdx.x * blockDim.x + threadIdx.x;
    const int    j   = (int)(idx & 63);
    const size_t row = idx >> 6;                    // bh * S + s
    const int    s   = (int)(row & (size_t)(S_ - 1));

    const float inv = __expf((float)j * (-2.0f * 9.210340371976184f / (float)DH_));
    float sn, cs;
    __sincosf((float)s * inv, &sn, &cs);

    const size_t base  = row * DH_;
    const float  scale = 0.08838834764831845f;      // 1/sqrt(128)

    const float q0 = bf2f(qd[base + j]);
    const float q1 = bf2f(qd[base + j + 64]);
    qd[base + j]      = f2bf((q0 * cs - q1 * sn) * scale);
    qd[base + j + 64] = f2bf((q1 * cs + q0 * sn) * scale);

    const float k0 = bf2f(kd[base + j]);
    const float k1 = bf2f(kd[base + j + 64]);
    kd[base + j]      = f2bf(k0 * cs - k1 * sn);
    kd[base + j + 64] = f2bf(k1 * cs + k0 * sn);
}

// ---------------------------------------------------------------------------
// Flash attention (causal, online softmax).
// Block = 128 threads = 4 waves = 4 consecutive q-tiles of ONE (b,h).
// The 32-key K tile (32x128) and V^T tile (128x32) are staged in double-
// buffered LDS by async global->LDS copies and shared by all 4 waves.
// Q stays in a per-wave LDS slot (round-2 spill fix); P transposes through a
// per-wave LDS slot.  All WMMA-containing branches are wave-uniform.
// LDS: Q 4x4KB + P 4x1KB + K 2x8KB + V 2x8KB = 52 KB.
// ---------------------------------------------------------------------------
__global__ void __launch_bounds__(128)
flash_attn_kernel(const u16* __restrict__ qd,
                  const u16* __restrict__ kd,
                  const u16* __restrict__ vtd,
                  u16* __restrict__ zb) {
    __shared__ u16 qlds[4][16 * 128];     // per-wave Q tile
    __shared__ u16 plds[4][16 * 32];      // per-wave P tile
    __shared__ u16 Kbuf[2][32 * 128];     // shared K tile  [key][dh]
    __shared__ u16 Vbuf[2][128 * 32];     // shared V^T tile [dh][key]

    const int tid  = threadIdx.x;
    const int lane = tid & 31;
    const int w    = tid >> 5;                  // wave 0..3
    const int bh   = blockIdx.x >> 5;           // 32 q-groups per (b,h)
    const int qt0  = (blockIdx.x & 31) * 4;
    const int qt   = qt0 + w;                   // this wave's q-tile
    const int b    = bh >> 4;                   // H_ == 16
    const int h    = bh & 15;
    const int lh   = lane >> 4, ll = lane & 15;

    const u16* kg_base = kd  + (size_t)bh * S_ * DH_;   // K  [key][dh]
    const u16* vg_base = vtd + (size_t)bh * DH_ * S_;   // V^T [dh][key]

    // ---- stage helper: copy K/V tiles for key-block `kb` into buffer `buf`.
    // K tile: 8KB contiguous. V tile: 128 rows of 64B, row stride S_ halves.
    // 512+512 16B chunks over 128 threads -> 8 async ops per thread.
    auto stage = [&](int kb, int buf) {
        const u16* kg = kg_base + (size_t)(kb * 32) * DH_;
        const u16* vg = vg_base + kb * 32;
#pragma unroll
        for (int i = 0; i < 4; ++i) {
            const int c = tid + i * 128;                 // 0..511
            cp16_g2l(&Kbuf[buf][c * 8], kg + c * 8);
        }
#pragma unroll
        for (int i = 0; i < 4; ++i) {
            const int c   = tid + i * 128;               // 0..511
            const int row = c >> 2;                      // dh row
            const int off = (c & 3) * 8;                 // halves within row
            cp16_g2l(&Vbuf[buf][row * 32 + off], vg + (size_t)row * S_ + off);
        }
        // prefetch one key-block further ahead (global_prefetch_b8 path)
        __builtin_prefetch(kg + (size_t)32 * DH_ + tid * 32, 0, 2);
    };

    // ---- per-wave Q tile -> LDS (wave-private, DS ops in-order per wave) ----
    {
        const u16* pq = qd + ((size_t)bh * S_ + qt * 16 + ll) * DH_ + lh * 64;
        u16*       pl = &qlds[w][ll * 128 + lh * 64];
#pragma unroll
        for (int i = 0; i < 4; ++i)
            *(uint4*)(pl + 16 * i) = *(const uint4*)(pq + 16 * i);
    }

    v8f   z[8] = {};
    float mr[8], lr[8];
#pragma unroll
    for (int j = 0; j < 8; ++j) { mr[j] = -3.0e38f; lr[j] = 0.0f; }

    const int nkb_mine = (qt * 16 + 47) >> 5;            // my causal blocks
    const int nkb_max  = ((qt0 + 3) * 16 + 47) >> 5;     // block-wide max

    stage(0, 0);
    int cur = 0;

    for (int kbi = 0; kbi < nkb_max; ++kbi) {
        const bool has_next = (kbi + 1 < nkb_max);
        if (has_next) {
            stage(kbi + 1, cur ^ 1);
            wait_async<8>();     // wait for the OLDER (cur) copies only
        } else {
            wait_async<0>();
        }
        __syncthreads();         // all waves' copies into `cur` visible

        if (kbi < nkb_mine) {    // wave-uniform causal skip
            // ---- scores S = Q * K^T (scale folded into Q) ----
            v8f s0 = {}, s1 = {};
#pragma unroll
            for (int i = 0; i < 4; ++i) {
                Frag16 a;
                {
                    const u16* pl = &qlds[w][ll * 128 + i * 32 + lh * 8];
                    a.q[0] = *(const uint4*)(pl);
                    a.q[1] = *(const uint4*)(pl + 16);
                }
                Frag16 b0, b1;
                const u16* pk0 = &Kbuf[cur][(ll)      * 128 + i * 32 + lh * 8];
                const u16* pk1 = &Kbuf[cur][(16 + ll) * 128 + i * 32 + lh * 8];
                b0.q[0] = *(const uint4*)(pk0);
                b0.q[1] = *(const uint4*)(pk0 + 16);
                b1.q[0] = *(const uint4*)(pk1);
                b1.q[1] = *(const uint4*)(pk1 + 16);
                s0 = __builtin_amdgcn_wmma_f32_16x16x32_bf16(
                    false, a.v, false, b0.v, (short)0, s0, false, false);
                s1 = __builtin_amdgcn_wmma_f32_16x16x32_bf16(
                    false, a.v, false, b1.v, (short)0, s1, false, false);
            }

            // ---- causal mask + online softmax ----
            const int c0 = kbi * 32 + ll;
#pragma unroll
            for (int j = 0; j < 8; ++j) {
                const int r = qt * 16 + j + lh * 8;
                float v0 = (c0      <= r) ? s0[j] : -3.0e38f;
                float v1 = (c0 + 16 <= r) ? s1[j] : -3.0e38f;

                float rm = fmaxf(v0, v1);
                rm = fmaxf(rm, __shfl_xor(rm, 1, 16));
                rm = fmaxf(rm, __shfl_xor(rm, 2, 16));
                rm = fmaxf(rm, __shfl_xor(rm, 4, 16));
                rm = fmaxf(rm, __shfl_xor(rm, 8, 16));

                const float mn    = fmaxf(mr[j], rm);
                const float alpha = __expf(mr[j] - mn);
                const float p0    = __expf(v0 - mn);
                const float p1    = __expf(v1 - mn);

                float rs = p0 + p1;
                rs += __shfl_xor(rs, 1, 16);
                rs += __shfl_xor(rs, 2, 16);
                rs += __shfl_xor(rs, 4, 16);
                rs += __shfl_xor(rs, 8, 16);

                lr[j] = lr[j] * alpha + rs;
                mr[j] = mn;
#pragma unroll
                for (int t = 0; t < 8; ++t) z[t][j] *= alpha;

                const int rr = j + lh * 8;
                plds[w][rr * 32 + ll]      = f2bf(p0);
                plds[w][rr * 32 + ll + 16] = f2bf(p1);
            }

            // ---- P A-fragment (transpose bounce, wave-private LDS) ----
            Frag16 pf;
            {
                const u16* pp = &plds[w][ll * 32 + lh * 8];
                pf.q[0] = *(const uint4*)(pp);
                pf.q[1] = *(const uint4*)(pp + 16);
            }

            // ---- z += P * V (V^T tile in LDS: contiguous key runs) ----
#pragma unroll
            for (int t = 0; t < 8; ++t) {
                const u16* pv = &Vbuf[cur][(t * 16 + ll) * 32 + lh * 8];
                Frag16 vf;
                vf.q[0] = *(const uint4*)(pv);
                vf.q[1] = *(const uint4*)(pv + 16);
                z[t] = __builtin_amdgcn_wmma_f32_16x16x32_bf16(
                    false, pf.v, false, vf.v, (short)0, z[t], false, false);
            }
        }

        __syncthreads();         // readers done before `cur` is overwritten
        cur ^= 1;
    }

    // ---- normalize and emit bf16 z laid out [B,S,D] ----
#pragma unroll
    for (int t = 0; t < 8; ++t) {
        const int d = h * DH_ + t * 16 + ll;
#pragma unroll
        for (int j = 0; j < 8; ++j) {
            const int r = qt * 16 + j + lh * 8;
            zb[((size_t)b * S_ + r) * D_ + d] = f2bf(z[t][j] / lr[j]);
        }
    }
}

// ---------------------------------------------------------------------------
// Launch
// ---------------------------------------------------------------------------
extern "C" void kernel_launch(void* const* d_in, const int* in_sizes, int n_in,
                              void* d_out, int out_size, void* d_ws, size_t ws_size,
                              hipStream_t stream) {
    (void)in_sizes; (void)n_in; (void)out_size; (void)ws_size;

    const float* x    = (const float*)d_in[0];
    const float* Wqkv = (const float*)d_in[1];
    const float* Wo   = (const float*)d_in[2];
    float*       out  = (float*)d_out;

    char* w = (char*)d_ws;
    auto take = [&](size_t bytes) -> char* {
        char* p = w;
        w += (bytes + 255) & ~(size_t)255;
        return p;
    };
    u16* xb    = (u16*)take((size_t)B_ * S_ * D_ * 2);
    u16* wqkvb = (u16*)take((size_t)3 * D_ * D_ * 2);
    u16* wob   = (u16*)take((size_t)D_ * D_ * 2);
    u16* qd    = (u16*)take((size_t)B_ * H_ * S_ * DH_ * 2);
    u16* kd    = (u16*)take((size_t)B_ * H_ * S_ * DH_ * 2);
    u16* vtd   = (u16*)take((size_t)B_ * H_ * DH_ * S_ * 2);
    u16* zb    = (u16*)take((size_t)B_ * S_ * D_ * 2);

    // 1) fp32 -> bf16
    f32_to_bf16_kernel<<<2048, 256, 0, stream>>>(x, xb, (size_t)B_ * S_ * D_);
    f32_to_bf16_kernel<<<2048, 256, 0, stream>>>(Wqkv, wqkvb, (size_t)3 * D_ * D_);
    f32_to_bf16_kernel<<<1024, 256, 0, stream>>>(Wo, wob, (size_t)D_ * D_);

    // 2) QKV projection GEMM + scatter epilogue
    {
        const int M = B_ * S_, N = 3 * D_, K = D_;
        const int waves = (M / 16) * (N / 64);
        gemm_bf16_kernel<1><<<waves / 8, 256, 0, stream>>>(
            xb, wqkvb, M, N, K, nullptr, qd, kd, vtd);
    }

    // 3) RoPE (+ q scale)
    rope_kernel<<<(B_ * H_ * S_ * 64) / 256, 256, 0, stream>>>(qd, kd);

    // 4) causal flash attention: 4 waves share one (b,h) K/V stream
    flash_attn_kernel<<<B_ * H_ * 32, 128, 0, stream>>>(qd, kd, vtd, zb);

    // 5) output projection GEMM -> fp32 out
    {
        const int M = B_ * S_, N = D_, K = D_;
        const int waves = (M / 16) * (N / 64);
        gemm_bf16_kernel<0><<<waves / 8, 256, 0, stream>>>(
            zb, wob, M, N, K, out, nullptr, nullptr, nullptr);
    }
}
